// GraphAttentionLayer_36112085025001
// MI455X (gfx1250) — matrix operand order repaired and checked by reading, not compiled, not run on hardware
//
#include <hip/hip_runtime.h>

typedef __attribute__((ext_vector_type(16))) _Float16 v16h;
typedef __attribute__((ext_vector_type(8)))  _Float16 v8h;
typedef __attribute__((ext_vector_type(8)))  float    v8f;

#define NB 8
#define NN 4096
#define NC 64
#define ND 64
#define NW (NN / 32)          /* mask words per row = 128 */
#define NEG_INF (-1e16f)

static __device__ __forceinline__ v8f wmma_f16(v16h a, v16h b, v8f c) {
  // D = A(16x32 f16) * B(32x16 f16) + C(16x16 f32)
  return __builtin_amdgcn_wmma_f32_16x16x32_f16(
      /*neg_a=*/false, a, /*neg_b=*/false, b,
      /*c_mod=*/(short)0, c, /*reuse_a=*/false, /*reuse_b=*/false);
}

// Async 16-byte memory->LDS copy per lane (CDNA5 async path, ASYNCcnt).
static __device__ __forceinline__ void async_copy_b128(const void* gptr, void* lptr) {
  unsigned lds           = (unsigned)(uintptr_t)lptr;        // LDS byte offset
  unsigned long long gaddr = (unsigned long long)(uintptr_t)gptr;
  asm volatile("global_load_async_to_lds_b128 %0, %1, off"
               :: "v"(lds), "v"(gaddr) : "memory");
}
static __device__ __forceinline__ void wait_async0() {
  asm volatile("s_wait_asynccnt 0x0" ::: "memory");
}

// ---------------------------------------------------------------------------
// Kernel 1: h16[b][n][d] = f16( sum_c x[b][n][c] * W[d][c] )
// One wave per 16-row tile. 8 WMMAs per wave.
// ---------------------------------------------------------------------------
__global__ void __launch_bounds__(256)
gat_h_kernel(const float* __restrict__ x, const float* __restrict__ W,
             _Float16* __restrict__ h16) {
  const int lane = threadIdx.x & 31;
  const int wave = blockIdx.x * (blockDim.x >> 5) + (threadIdx.x >> 5);
  const int b    = wave >> 8;                // 256 tiles per batch
  const int row0 = (wave & 255) << 4;
  const int n    = lane & 15;                // A row m == B col n == lane%16
  const int half = lane >> 4;

  // A operand: rows of x; element e -> k=16h+e  (+32 for chunk 1)
  const float* xr = x + ((size_t)b * NN + row0 + n) * NC;
  v16h a0, a1;
#pragma unroll
  for (int e = 0; e < 8; ++e) {
    a0[e]     = (_Float16)xr[16 * half + e];
    a0[8 + e] = (_Float16)xr[8 + 16 * half + e];
    a1[e]     = (_Float16)xr[32 + 16 * half + e];
    a1[8 + e] = (_Float16)xr[40 + 16 * half + e];
  }

#pragma unroll
  for (int dt = 0; dt < 4; ++dt) {
    // B operand: B[k][n] = W[dt*16+n][k]; element e -> k = 32*chunk + 16*half + e
    const float* wr = W + (size_t)(dt * 16 + n) * NC;
    v16h b0, b1;
#pragma unroll
    for (int e = 0; e < 16; ++e) {
      b0[e] = (_Float16)wr[16 * half + e];
      b1[e] = (_Float16)wr[32 + 16 * half + e];
    }
    v8f acc = {};
    acc = wmma_f16(a0, b0, acc);
    acc = wmma_f16(a1, b1, acc);
    // D layout: VGPR r -> row r+8*half, col = n
#pragma unroll
    for (int r = 0; r < 8; ++r) {
      h16[((size_t)b * NN + row0 + r + 8 * half) * ND + dt * 16 + n] =
          (_Float16)acc[r];
    }
  }
}

// ---------------------------------------------------------------------------
// Kernel 2: pack graph int32 {0,1} -> bitmask [N][N/32]
// ---------------------------------------------------------------------------
__global__ void __launch_bounds__(256)
gat_pack_kernel(const int* __restrict__ g, unsigned* __restrict__ mask) {
  const int t   = blockIdx.x * blockDim.x + threadIdx.x;  // N*NW threads exactly
  const int row = t >> 7;
  const int w   = t & (NW - 1);
  const int* gp = g + (size_t)row * NN + w * 32;
  unsigned bits = 0u;
#pragma unroll
  for (int i = 0; i < 32; ++i) bits |= (gp[i] != 0 ? 1u : 0u) << i;
  mask[t] = bits;
}

// ---------------------------------------------------------------------------
// Kernel 3: fused flash attention. 256 threads = 8 waves; each wave owns a
// 16-row Q tile; all waves share the same 32-key chunk, which is staged into
// LDS one iteration ahead with async memory->LDS loads (double buffered).
// ---------------------------------------------------------------------------
__global__ void __launch_bounds__(256)
gat_attn_kernel(const _Float16* __restrict__ h16,
                const unsigned* __restrict__ gmask,
                const float* __restrict__ bias,
                float* __restrict__ out) {
  __shared__ __align__(32) _Float16 kv[2][32 * ND];    // staged K/V chunks: 2x4 KB
  __shared__ __align__(32) _Float16 vT[ND * 32];       // V^T [d][j]        : 4 KB
  __shared__ __align__(16) _Float16 pbuf[8][16 * 32];  // per-wave P tile   : 8 KB

  const int t    = threadIdx.x;
  const int lane = t & 31;
  const int w    = t >> 5;
  const int b    = blockIdx.x >> 5;                        // 32 blocks per batch
  const int row0 = (((blockIdx.x & 31) << 3) | w) << 4;    // 16-row Q tile
  const int n    = lane & 15;
  const int half = lane >> 4;

  const _Float16* hb = h16 + (size_t)b * NN * ND;

  // Q A-operand (two K=32 chunks), element e -> k = 16*half + e
  const _Float16* qr = hb + (size_t)(row0 + n) * ND;
  v8h q00 = *(const v8h*)(qr + 16 * half);
  v8h q01 = *(const v8h*)(qr + 8 + 16 * half);
  v8h q10 = *(const v8h*)(qr + 32 + 16 * half);
  v8h q11 = *(const v8h*)(qr + 40 + 16 * half);
  v16h a0 = __builtin_shufflevector(q00, q01, 0,1,2,3,4,5,6,7,8,9,10,11,12,13,14,15);
  v16h a1 = __builtin_shufflevector(q10, q11, 0,1,2,3,4,5,6,7,8,9,10,11,12,13,14,15);

  v8f Oa[4] = {v8f{}, v8f{}, v8f{}, v8f{}};
  float mrow[8], lrow[8];
#pragma unroll
  for (int r = 0; r < 8; ++r) { mrow[r] = -3.0e38f; lrow[r] = 0.0f; }

  _Float16* pw = &pbuf[w][0];

  // Prologue: async-stage chunk 0 (each thread copies 16 bytes).
  async_copy_b128(hb + (size_t)t * 8, &kv[0][t * 8]);

  for (int jc = 0; jc < NW; ++jc) {
    const int cur = jc & 1;

    wait_async0();       // this wave's staged slice has landed in LDS
    __syncthreads();     // -> kv[cur] complete and visible to all waves

    // Issue async stage of next chunk into the other buffer (overlaps compute).
    if (jc + 1 < NW) {
      async_copy_b128(hb + (size_t)(jc + 1) * 32 * ND + (size_t)t * 8,
                      &kv[1 - cur][t * 8]);
    }

    // Transpose pass: kv[cur] (row-major [j][d]) -> vT ([d][j])
    {
      const int j  = t >> 3;
      const int d0 = (t & 7) << 3;
      v8h row = *(const v8h*)&kv[cur][j * ND + d0];
#pragma unroll
      for (int e = 0; e < 8; ++e) vT[(d0 + e) * 32 + j] = row[e];
    }

    // Score-phase B operands from staged LDS.
    // B[k][col]: lane col = n, element e -> k = 32*chunk + 16*half + e
    const _Float16* krow0 = &kv[cur][n * ND];          // key j0+n
    const _Float16* krow1 = &kv[cur][(n + 16) * ND];   // key j0+16+n
    v16h kb00 = *(const v16h*)(krow0 + 16 * half);
    v16h kb01 = *(const v16h*)(krow0 + 32 + 16 * half);
    v16h kb10 = *(const v16h*)(krow1 + 16 * half);
    v16h kb11 = *(const v16h*)(krow1 + 32 + 16 * half);

    v8f s0 = {}, s1 = {};
    s0 = wmma_f16(a0, kb00, s0);
    s0 = wmma_f16(a1, kb01, s0);
    s1 = wmma_f16(a0, kb10, s1);
    s1 = wmma_f16(a1, kb11, s1);

    // Mask + online softmax. D-layout: VGPR r -> row r+8*half, lane -> col n.
#pragma unroll
    for (int r = 0; r < 8; ++r) {
      const unsigned bits = gmask[(size_t)(row0 + r + 8 * half) * NW + jc];
      const float v0 = ((bits >> n) & 1u)        ? s0[r] : NEG_INF;
      const float v1 = ((bits >> (n + 16)) & 1u) ? s1[r] : NEG_INF;
      float mx = fmaxf(v0, v1);
      mx = fmaxf(mx, __shfl_xor(mx, 1, 32));
      mx = fmaxf(mx, __shfl_xor(mx, 2, 32));
      mx = fmaxf(mx, __shfl_xor(mx, 4, 32));
      mx = fmaxf(mx, __shfl_xor(mx, 8, 32));
      const float mn   = fmaxf(mrow[r], mx);
      const float corr = __expf(mrow[r] - mn);
      mrow[r] = mn;
      const float p0 = __expf(v0 - mn);
      const float p1 = __expf(v1 - mn);
      float rs = p0 + p1;
      rs += __shfl_xor(rs, 1, 32);
      rs += __shfl_xor(rs, 2, 32);
      rs += __shfl_xor(rs, 4, 32);
      rs += __shfl_xor(rs, 8, 32);
      lrow[r] = lrow[r] * corr + rs;
      Oa[0][r] *= corr; Oa[1][r] *= corr; Oa[2][r] *= corr; Oa[3][r] *= corr;
      pw[(r + 8 * half) * 32 + n]      = (_Float16)p0;   // P tile, row-major
      pw[(r + 8 * half) * 32 + n + 16] = (_Float16)p1;
    }

    __syncthreads();  // vT fully transposed (also fences own DS ops)

    // P as A-operand: row m=n, element e -> k=16*half+e
    v8h pl0 = *(const v8h*)(pw + n * 32 + 16 * half);
    v8h pl1 = *(const v8h*)(pw + n * 32 + 8 + 16 * half);
    v16h pa = __builtin_shufflevector(pl0, pl1, 0,1,2,3,4,5,6,7,8,9,10,11,12,13,14,15);

    // V as B-operand from transposed LDS: element e -> k = 16*half + e
#pragma unroll
    for (int dt = 0; dt < 4; ++dt) {
      v16h vb = *(const v16h*)(&vT[(dt * 16 + n) * 32 + 16 * half]);
      Oa[dt] = wmma_f16(pa, vb, Oa[dt]);
    }
  }

  // Epilogue: divide by softmax denom, add bias, store fp32.
  float* orow = out + ((size_t)b * NN + row0) * ND;
#pragma unroll
  for (int dt = 0; dt < 4; ++dt) {
    const float bb = bias[dt * 16 + n];
#pragma unroll
    for (int r = 0; r < 8; ++r) {
      orow[(size_t)(r + 8 * half) * ND + dt * 16 + n] =
          Oa[dt][r] / lrow[r] + bb;
    }
  }
}

// ---------------------------------------------------------------------------
extern "C" void kernel_launch(void* const* d_in, const int* in_sizes, int n_in,
                              void* d_out, int out_size, void* d_ws, size_t ws_size,
                              hipStream_t stream) {
  (void)in_sizes; (void)n_in; (void)out_size; (void)ws_size;
  const float* x     = (const float*)d_in[0];   // [B,N,C] f32
  const int*   graph = (const int*)  d_in[1];   // [N,N]  i32
  const float* W     = (const float*)d_in[2];   // [D,C]  f32
  const float* bias  = (const float*)d_in[3];   // [D]    f32
  float*       out   = (float*)d_out;           // [B,N,D] f32

  _Float16* h16   = (_Float16*)d_ws;                                       // 4 MB
  unsigned* gmask = (unsigned*)((char*)d_ws +
                                (size_t)NB * NN * ND * sizeof(_Float16));  // 2 MB

  gat_h_kernel   <<<NB * NN / 16 / 8, 256, 0, stream>>>(x, W, h16);
  gat_pack_kernel<<<NN * NW / 256,    256, 0, stream>>>(graph, gmask);
  gat_attn_kernel<<<NB * (NN / 128),  256, 0, stream>>>(h16, gmask, bias, out);
}
